// SingleAttention_42133629174493
// MI455X (gfx1250) — matrix operand order repaired and checked by reading, not compiled, not run on hardware
//
#include <hip/hip_runtime.h>

// -----------------------------------------------------------------------------
// Fused single-head attention for MI455X (gfx1250, wave32, WMMA).
//
// Roofline: inputs are 100 MB fp32 (q_in/k_in/v_in) -> projection pass is
// HBM-bound (~4.3us @ 23.3 TB/s); all three projections run in ONE launch
// (gridDim.y=3) and are software-pipelined (double-buffered LDS) so global
// loads stay in flight under the WMMAs. Projected q/k/v are 6 MB f16 -> the
// attention phase runs out of the 192 MB L2. Each attention block stages the
// shared 32-key K/V tile in LDS once for its 8 waves (8x less L2 traffic),
// double-buffered, with V stored transposed so WMMA B-fragments are
// contiguous b128 LDS loads. Softmax scale folds 0.125*log2(e) into q so the
// online softmax uses raw v_exp_f32 (exp2).
// -----------------------------------------------------------------------------

typedef __attribute__((ext_vector_type(16))) _Float16 v16h;
typedef __attribute__((ext_vector_type(8)))  _Float16 v8h;
typedef __attribute__((ext_vector_type(8)))  float    v8f;
typedef __attribute__((ext_vector_type(4)))  float    v4f;

#define NB    8
#define SEQ   2048
#define DIN   512
#define DKV   64
#define NROWS (NB * SEQ)   // 16384 rows total
#define LOG2E 1.44269504088896340736f

static __device__ __forceinline__ v8f wmma_f16(v16h a, v16h b, v8f c) {
  // D = A(16x32 f16) * B(32x16 f16) + C(16x16 f32)
  return __builtin_amdgcn_wmma_f32_16x16x32_f16(
      /*neg_a=*/false, a, /*neg_b=*/false, b,
      /*c_mod=*/(short)0, c, /*reuse_a=*/false, /*reuse_b=*/false);
}

static __device__ __forceinline__ v16h cat8(v8h lo, v8h hi) {
  return __builtin_shufflevector(lo, hi, 0,1,2,3,4,5,6,7,8,9,10,11,12,13,14,15);
}

// =============================================================================
// Projections (all three in one launch, blockIdx.y selects q/k/v):
//   Y[NROWS x 64] = f16( (X[NROWS x 512] @ W[512 x 64] + bias) * scale )
// Block = 256 threads (8 waves); 128-row panel per block; each wave owns a
// 16-row tile and all four 16-col output tiles. Double-buffered LDS pipeline:
// load chunk c+1 to VGPRs -> compute chunk c -> store chunk c+1 -> barrier.
// =============================================================================
__global__ __launch_bounds__(256) void proj_wmma(
    const float* __restrict__ Xq, const float* __restrict__ Xk, const float* __restrict__ Xv,
    const float* __restrict__ Wq, const float* __restrict__ Wk, const float* __restrict__ Wv,
    const float* __restrict__ bq, const float* __restrict__ bk, const float* __restrict__ bv,
    _Float16* __restrict__ Yq, _Float16* __restrict__ Yk, _Float16* __restrict__ Yv)
{
  __shared__ __align__(16) _Float16 lx [2][128][40];  // X chunk, 80B stride
  __shared__ __align__(16) _Float16 lwT[2][64][40];   // W chunk transposed [n][k]

  const int which = blockIdx.y;            // 0=q 1=k 2=v
  const float* __restrict__ X    = (which == 0) ? Xq : (which == 1) ? Xk : Xv;
  const float* __restrict__ W    = (which == 0) ? Wq : (which == 1) ? Wk : Wv;
  const float* __restrict__ bias = (which == 0) ? bq : (which == 1) ? bk : bv;
  _Float16* __restrict__ Y       = (which == 0) ? Yq : (which == 1) ? Yk : Yv;
  // fold 1/sqrt(d_k) and log2(e) (for exp2-based softmax) into q
  const float outScale = (which == 0) ? 0.125f * LOG2E : 1.0f;

  const int tid    = threadIdx.x;
  const int lane   = tid & 31;
  const int wave   = tid >> 5;
  const int half16 = lane >> 4;
  const int l16    = lane & 15;
  const int rowPanel = blockIdx.x * 128;

  // staging indices: X via float4 (8 threads/row), W via 8 scalar floats
  const int r4 = tid >> 3;          // 0..31
  const int c4 = (tid & 7) * 4;     // 0,4,...,28

  const v8f zero = {0.f,0.f,0.f,0.f,0.f,0.f,0.f,0.f};
  v8f acc[4] = {zero, zero, zero, zero};

  v4f  xs[4];
  float ws[8];

  // ---- prologue: stage chunk 0 into buffer 0 ----
  #pragma unroll
  for (int i = 0; i < 4; ++i)
    xs[i] = *(const v4f*)(X + (size_t)(rowPanel + r4 + i * 32) * DIN + c4);
  #pragma unroll
  for (int i = 0; i < 8; ++i) {
    const int idx = i * 256 + tid;
    ws[i] = W[(size_t)(idx >> 6) * DKV + (idx & 63)];
  }
  #pragma unroll
  for (int i = 0; i < 4; ++i)
    #pragma unroll
    for (int u = 0; u < 4; ++u) lx[0][r4 + i * 32][c4 + u] = (_Float16)xs[i][u];
  #pragma unroll
  for (int i = 0; i < 8; ++i) {
    const int idx = i * 256 + tid;
    lwT[0][idx & 63][idx >> 6] = (_Float16)ws[i];
  }
  __syncthreads();

  const int arow = wave * 16 + l16;
  const int kb   = half16 * 8;
  const int kb2  = half16 * 16;

  for (int c = 0; c < DIN / 32; ++c) {
    const int cur = c & 1;
    const int k1  = (c + 1) * 32;

    // ---- issue next chunk's global loads (in flight under the WMMAs) ----
    if (k1 < DIN) {
      #pragma unroll
      for (int i = 0; i < 4; ++i) {
        const float* src = X + (size_t)(rowPanel + r4 + i * 32) * DIN + k1 + c4;
        xs[i] = *(const v4f*)src;
        __builtin_prefetch(src + 32, 0, 0);   // chunk c+2 -> global_prefetch
      }
      #pragma unroll
      for (int i = 0; i < 8; ++i) {
        const int idx = i * 256 + tid;
        ws[i] = W[(size_t)(k1 + (idx >> 6)) * DKV + (idx & 63)];
      }
    }

    // ---- compute chunk c: hoist all fragments, then 4 WMMAs ----
    const v16h a = cat8(*(const v8h*)&lx[cur][arow][kb2],
                        *(const v8h*)&lx[cur][arow][kb2 + 8]);
    v16h bfr[4];
    #pragma unroll
    for (int nt = 0; nt < 4; ++nt) {
      const int n = nt * 16 + l16;
      bfr[nt] = cat8(*(const v8h*)&lwT[cur][n][kb2],
                     *(const v8h*)&lwT[cur][n][kb2 + 8]);
    }
    #pragma unroll
    for (int nt = 0; nt < 4; ++nt) acc[nt] = wmma_f16(a, bfr[nt], acc[nt]);

    // ---- store next chunk into the other buffer ----
    if (k1 < DIN) {
      const int nxt = cur ^ 1;
      #pragma unroll
      for (int i = 0; i < 4; ++i)
        #pragma unroll
        for (int u = 0; u < 4; ++u) lx[nxt][r4 + i * 32][c4 + u] = (_Float16)xs[i][u];
      #pragma unroll
      for (int i = 0; i < 8; ++i) {
        const int idx = i * 256 + tid;
        lwT[nxt][idx & 63][idx >> 6] = (_Float16)ws[i];
      }
    }
    __syncthreads();
  }

  // ---- epilogue: bias, scale, f16 store (C layout: VGPR r -> row r+8*half) ----
  const int m0 = rowPanel + wave * 16;
  #pragma unroll
  for (int nt = 0; nt < 4; ++nt) {
    const int n = nt * 16 + l16;
    const float bb = bias[n];
    #pragma unroll
    for (int r = 0; r < 8; ++r) {
      const int row = m0 + r + half16 * 8;
      Y[(size_t)row * DKV + n] = (_Float16)((acc[nt][r] + bb) * outScale);
    }
  }
}

// =============================================================================
// Flash attention: block = 8 waves covering 128 consecutive queries of one
// batch; shared 32-key K/V tile double-buffered in LDS (V transposed for
// contiguous B-fragments). Per wave: online softmax (exp2-based) with
// shfl_xor row reductions; WMMA for QK^T and P*V. q carries 0.125*log2e.
// =============================================================================
__global__ __launch_bounds__(256) void attn_wmma(
    const _Float16* __restrict__ Q, const _Float16* __restrict__ K,
    const _Float16* __restrict__ V, float* __restrict__ Out)
{
  __shared__ __align__(16) _Float16 lk [2][32][72];    // K tile [key][dk]
  __shared__ __align__(16) _Float16 lvT[2][64][40];    // V tile transposed [n][key]
  __shared__ __align__(16) _Float16 lp [8][16][32];    // per-wave P tile

  const int tid  = threadIdx.x;
  const int lane = tid & 31;
  const int wave = tid >> 5;
  const int half16 = lane >> 4;
  const int l16    = lane & 15;

  const int b  = blockIdx.x >> 4;                  // 16 blocks per batch
  const int q0 = (blockIdx.x & 15) * 128 + wave * 16;

  // --- Q A-fragments for dk 0..31 and 32..63 (loaded once) ---
  const size_t qrow = (size_t)(b * SEQ + q0 + l16) * DKV;
  const int kb = half16 * 8;
  v16h aq0, aq1;
  #pragma unroll
  for (int e = 0; e < 16; ++e) {
    const int kk = half16 * 16 + e;
    aq0[e] = Q[qrow + kk];
    aq1[e] = Q[qrow + 32 + kk];
  }

  const v8f zero = {0.f,0.f,0.f,0.f,0.f,0.f,0.f,0.f};
  v8f acc[4] = {zero, zero, zero, zero};
  float mstat[8], lstat[8];
  #pragma unroll
  for (int r = 0; r < 8; ++r) { mstat[r] = -1e30f; lstat[r] = 0.0f; }

  const int kb2  = half16 * 16;
  const int koff = half16 * 16;
  const size_t kvbase = (size_t)b * SEQ * DKV;

  // cooperative staging: one v8h of K and V per thread
  const int skey = tid >> 3;          // 0..31
  const int sc0  = (tid & 7) * 8;     // 0,8,...,56

  // ---- prologue: stage tile 0 into buffer 0 ----
  {
    const size_t g = kvbase + (size_t)skey * DKV + sc0;
    const v8h k8 = *(const v8h*)(K + g);
    const v8h v8 = *(const v8h*)(V + g);
    *(v8h*)&lk[0][skey][sc0] = k8;
    #pragma unroll
    for (int u = 0; u < 8; ++u) lvT[0][sc0 + u][skey] = v8[u];
  }
  __syncthreads();

  for (int kt = 0; kt < SEQ / 32; ++kt) {
    const int cur  = kt & 1;
    const bool more = (kt + 1) < SEQ / 32;

    // ---- issue next tile's global loads (hidden under this tile's work) ----
    v8h k8{}, v8n{};
    if (more) {
      const size_t g = kvbase + (size_t)((kt + 1) * 32 + skey) * DKV + sc0;
      k8  = *(const v8h*)(K + g);
      v8n = *(const v8h*)(V + g);
    }

    // ---- scores: hoist 4 B-fragments, then 4 WMMAs ----
    const _Float16* kr0 = &lk[cur][l16][0];       // sub-tile j=0
    const _Float16* kr1 = &lk[cur][16 + l16][0];  // sub-tile j=1
    const v16h bk00 = cat8(*(const v8h*)(kr0 + kb2),      *(const v8h*)(kr0 + kb2 + 8));
    const v16h bk01 = cat8(*(const v8h*)(kr0 + 32 + kb2), *(const v8h*)(kr0 + 32 + kb2 + 8));
    const v16h bk10 = cat8(*(const v8h*)(kr1 + kb2),      *(const v8h*)(kr1 + kb2 + 8));
    const v16h bk11 = cat8(*(const v8h*)(kr1 + 32 + kb2), *(const v8h*)(kr1 + 32 + kb2 + 8));
    v8f s[2];
    s[0] = wmma_f16(aq0, bk00, zero);
    s[0] = wmma_f16(aq1, bk01, s[0]);
    s[1] = wmma_f16(aq0, bk10, zero);
    s[1] = wmma_f16(aq1, bk11, s[1]);

    // ---- P*V B-fragments loaded early: DS latency hides under softmax ----
    v16h bv[4];
    #pragma unroll
    for (int nt = 0; nt < 4; ++nt) {
      const int n = nt * 16 + l16;
      bv[nt] = cat8(*(const v8h*)&lvT[cur][n][koff],
                    *(const v8h*)&lvT[cur][n][koff + 8]);
    }

    // ---- online softmax over the 32 new keys (base-2, rows in 16-lane grps) ----
    float corr[8];
    #pragma unroll
    for (int r = 0; r < 8; ++r) {
      float v = fmaxf(s[0][r], s[1][r]);
      v = fmaxf(v, __shfl_xor(v, 1, 16));
      v = fmaxf(v, __shfl_xor(v, 2, 16));
      v = fmaxf(v, __shfl_xor(v, 4, 16));
      v = fmaxf(v, __shfl_xor(v, 8, 16));
      const float mn = fmaxf(mstat[r], v);
      corr[r] = exp2f(mstat[r] - mn);
      mstat[r] = mn;
      const float p0 = exp2f(s[0][r] - mn);
      const float p1 = exp2f(s[1][r] - mn);
      float rs = p0 + p1;
      rs += __shfl_xor(rs, 1, 16);
      rs += __shfl_xor(rs, 2, 16);
      rs += __shfl_xor(rs, 4, 16);
      rs += __shfl_xor(rs, 8, 16);
      lstat[r] = lstat[r] * corr[r] + rs;
      // stage P (f16) into per-wave LDS tile; rows r+8*half, cols l16/16+l16
      lp[wave][r + half16 * 8][l16]      = (_Float16)p0;
      lp[wave][r + half16 * 8][16 + l16] = (_Float16)p1;
    }
    #pragma unroll
    for (int nt = 0; nt < 4; ++nt)
      #pragma unroll
      for (int r = 0; r < 8; ++r) acc[nt][r] = acc[nt][r] * corr[r];

    // ---- P A-fragment (16x32) from LDS (same-wave DS ops are in-order) ----
    const v16h pa = cat8(*(const v8h*)&lp[wave][l16][kb2],
                         *(const v8h*)&lp[wave][l16][kb2 + 8]);

    #pragma unroll
    for (int nt = 0; nt < 4; ++nt) acc[nt] = wmma_f16(pa, bv[nt], acc[nt]);

    // ---- store next tile into the other buffer, single barrier per iter ----
    if (more) {
      const int nxt = cur ^ 1;
      *(v8h*)&lk[nxt][skey][sc0] = k8;
      #pragma unroll
      for (int u = 0; u < 8; ++u) lvT[nxt][sc0 + u][skey] = v8n[u];
    }
    __syncthreads();
  }

  // ---- normalize by row sums and store fp32 output ----
  #pragma unroll
  for (int nt = 0; nt < 4; ++nt) {
    const int n = nt * 16 + l16;
    #pragma unroll
    for (int r = 0; r < 8; ++r) {
      const int row = q0 + r + half16 * 8;
      Out[((size_t)b * SEQ + row) * DKV + n] = acc[nt][r] / lstat[r];
    }
  }
}

// =============================================================================
extern "C" void kernel_launch(void* const* d_in, const int* in_sizes, int n_in,
                              void* d_out, int out_size, void* d_ws, size_t ws_size,
                              hipStream_t stream) {
  const float* q_in = (const float*)d_in[0];
  const float* k_in = (const float*)d_in[1];
  const float* v_in = (const float*)d_in[2];
  const float* Wq   = (const float*)d_in[3];
  const float* bq   = (const float*)d_in[4];
  const float* Wk   = (const float*)d_in[5];
  const float* bk   = (const float*)d_in[6];
  const float* Wv   = (const float*)d_in[7];
  const float* bv   = (const float*)d_in[8];

  // workspace: f16 q/k/v, 2 MB each (6 MB total)
  _Float16* qws = (_Float16*)d_ws;
  _Float16* kws = qws + (size_t)NROWS * DKV;
  _Float16* vws = kws + (size_t)NROWS * DKV;

  // all three projections in one launch: 128 x 3 workgroups streaming HBM
  proj_wmma<<<dim3(NROWS / 128, 3), dim3(256), 0, stream>>>(
      q_in, k_in, v_in, Wq, Wk, Wv, bq, bk, bv, qws, kws, vws);

  // 8 batches x 16 blocks; each block = 8 waves x 16-query tiles
  attn_wmma<<<dim3(NB * 16), dim3(256), 0, stream>>>(qws, kws, vws, (float*)d_out);
}
